// HHDocGraphSum_5574867550778
// MI455X (gfx1250) — matrix operand order repaired and checked by reading, not compiled, not run on hardware
//
#include <hip/hip_runtime.h>
#include <math.h>

// ---------------- fixed problem sizes (from reference) ----------------
#define BB     16
#define SS_    128
#define NS     (BB * SS_)      // 2048 sentences
#define NW     50000           // word nodes
#define VD     120             // word emb dim D
#define NF     64
#define HH     64
#define FE     64              // feat embed size F
#define FFN    512
#define EWS    500000
#define EWW    300000
#define ESS    100000

typedef _Float16 h16 __attribute__((ext_vector_type(16)));
typedef float    f8  __attribute__((ext_vector_type(8)));

static inline int cdiv(int a, int b) { return (a + b - 1) / b; }

// =====================================================================
//  WMMA GEMM:  C = act(A[M,K] @ B[K,N] + bias + residual)
//  One wave32 per 16(M) x 32(N) C tile, K stepped by 32.
//  B tiles double-buffered in LDS via global_load_async_to_lds_b128
//  (ASYNCcnt software pipeline); A streamed with float2 loads.
//  All our M are multiples of 16 and K is even -> branch-free inner loop
//  (clamped addresses; out-of-range N columns computed but never stored).
//  act: 0=none 1=relu 2=sigmoid
// =====================================================================

#define BUILD_A_FULL(k0)                                                     \
    do {                                                                     \
        _Pragma("unroll")                                                    \
        for (int v = 0; v < 8; ++v) {                                        \
            int k = (k0) + ((v < 4) ? 0 : 16) + hi * 8 + 2 * (v & 3);        \
            float2 p = *(const float2*)(arow + k);                           \
            a[2 * v]     = (_Float16)p.x;                                    \
            a[2 * v + 1] = (_Float16)p.y;                                    \
        }                                                                    \
    } while (0)

#define BUILD_A_TAIL(k0)                                                     \
    do {                                                                     \
        _Pragma("unroll")                                                    \
        for (int v = 0; v < 8; ++v) {                                        \
            int k  = (k0) + ((v < 4) ? 0 : 16) + hi * 8 + 2 * (v & 3);       \
            int kc = (k > K - 2) ? (K - 2) : k;                              \
            float2 p = *(const float2*)(arow + kc);                          \
            a[2 * v]     = (k     < K) ? (_Float16)p.x : (_Float16)0.f;      \
            a[2 * v + 1] = (k + 1 < K) ? (_Float16)p.y : (_Float16)0.f;      \
        }                                                                    \
    } while (0)

#define BUILD_B(bufi)                                                        \
    do {                                                                     \
        _Pragma("unroll")                                                    \
        for (int v = 0; v < 8; ++v) {                                        \
            int r = ((v < 4) ? 0 : 16) + hi * 8 + 2 * (v & 3);              \
            b0[2 * v]     = (_Float16)bt[bufi][r * 32 + l15];                \
            b0[2 * v + 1] = (_Float16)bt[bufi][(r + 1) * 32 + l15];          \
            b1[2 * v]     = (_Float16)bt[bufi][r * 32 + 16 + l15];           \
            b1[2 * v + 1] = (_Float16)bt[bufi][(r + 1) * 32 + 16 + l15];     \
        }                                                                    \
    } while (0)

#define DO_WMMA()                                                            \
    do {                                                                     \
        acc0 = __builtin_amdgcn_wmma_f32_16x16x32_f16(false, a, false, b0,   \
                                                      (short)0, acc0, false, false); \
        acc1 = __builtin_amdgcn_wmma_f32_16x16x32_f16(false, a, false, b1,   \
                                                      (short)0, acc1, false, false); \
    } while (0)

#define STAGE_ASYNC(k0, bufi)                                                \
    do {                                                                     \
        const float* gp = B + (size_t)((k0) + lane) * N + cb;                \
        unsigned lp = (unsigned)(unsigned long long)                         \
            (__attribute__((address_space(3))) float*)&bt[bufi][lane * 32];  \
        _Pragma("unroll")                                                    \
        for (int j = 0; j < 8; ++j)                                          \
            asm volatile("global_load_async_to_lds_b128 %0, %1, off"         \
                         :: "v"(lp + 16u * j),                               \
                            "v"((unsigned long long)(size_t)(gp + 4 * j))    \
                         : "memory");                                        \
    } while (0)

__global__ __launch_bounds__(32)
void k_gemm_wmma(const float* __restrict__ A, const float* __restrict__ B,
                 const float* __restrict__ bias, const float* __restrict__ res,
                 float* __restrict__ C, int M, int N, int K, int act)
{
    __shared__ __align__(16) float bt[2][32 * 32];
    const int tm = blockIdx.x, tn = blockIdx.y;
    const int lane = threadIdx.x, l15 = lane & 15, hi = lane >> 4;
    int mrow = tm * 16 + l15;
    if (mrow > M - 1) mrow = M - 1;          // M is a multiple of 16 in all calls
    const float* arow = A + (size_t)mrow * K;
    const int cb = tn * 32;
    const bool fastN = (cb + 32 <= N) && ((N & 3) == 0);
    const int nfull = K >> 5;

    f8 acc0 = {}, acc1 = {};
    h16 a, b0, b1;

    if (fastN && nfull > 0) {
        STAGE_ASYNC(0, 0);
        for (int it = 0; it < nfull; ++it) {
            const int k0 = it * 32;
            const int buf = it & 1;
            if (it + 1 < nfull) {
                STAGE_ASYNC(k0 + 32, buf ^ 1);
                asm volatile("s_wait_asynccnt 8" ::: "memory");
            } else {
                asm volatile("s_wait_asynccnt 0" ::: "memory");
            }
            if (k0 + 32 < K)
                __builtin_prefetch(arow + k0 + 32, 0, 0);   // global_prefetch_b8
            BUILD_A_FULL(k0);
            BUILD_B(buf);
            DO_WMMA();
        }
    } else {
        for (int it = 0; it < nfull; ++it) {
            const int k0 = it * 32;
            for (int idx = lane; idx < 1024; idx += 32) {
                int r = idx >> 5, c = idx & 31;
                int cc = cb + c;
                if (cc > N - 1) cc = N - 1;   // garbage cols never stored
                bt[0][idx] = B[(size_t)(k0 + r) * N + cc];
            }
            BUILD_A_FULL(k0);
            BUILD_B(0);
            DO_WMMA();
        }
    }
    // ---- K tail (guarded by clamped addresses + selects, A zero-filled) ----
    const int kt = nfull << 5;
    if (kt < K) {
        for (int idx = lane; idx < 1024; idx += 32) {
            int r = idx >> 5, c = idx & 31;
            int kk = kt + r;
            if (kk > K - 1) kk = K - 1;       // A tail is zero there -> product 0
            int cc = cb + c;
            if (cc > N - 1) cc = N - 1;
            bt[0][idx] = B[(size_t)kk * N + cc];
        }
        BUILD_A_TAIL(kt);
        BUILD_B(0);
        DO_WMMA();
    }

    // ---- epilogue: D VGPR r holds M = r + hi*8, N = lane&15 ----
#pragma unroll
    for (int r = 0; r < 8; ++r) {
        int row = tm * 16 + r + hi * 8;
        if (row < M) {
            int c0 = cb + l15, c1 = cb + 16 + l15;
            if (c0 < N) {
                float v = acc0[r];
                if (bias) v += bias[c0];
                if (res)  v += res[(size_t)row * N + c0];
                if (act == 1)      v = fmaxf(v, 0.f);
                else if (act == 2) v = 1.f / (1.f + __expf(-v));
                C[(size_t)row * N + c0] = v;
            }
            if (c1 < N) {
                float v = acc1[r];
                if (bias) v += bias[c1];
                if (res)  v += res[(size_t)row * N + c1];
                if (act == 1)      v = fmaxf(v, 0.f);
                else if (act == 2) v = 1.f / (1.f + __expf(-v));
                C[(size_t)row * N + c1] = v;
            }
        }
    }
}

// =====================================================================
//  Elementwise / gather / segment kernels
// =====================================================================
__global__ void k_fill(float* p, long long n, float v) {
    for (long long i = blockIdx.x * (long long)blockDim.x + threadIdx.x;
         i < n; i += (long long)gridDim.x * blockDim.x) p[i] = v;
}

__global__ void k_gather_rows(const float* __restrict__ tab, const int* __restrict__ idx,
                              float* __restrict__ out, long long n, int dim) {
    for (long long t = blockIdx.x * (long long)blockDim.x + threadIdx.x;
         t < n * dim; t += (long long)gridDim.x * blockDim.x) {
        long long i = t / dim; int d = (int)(t % dim);
        out[t] = tab[(size_t)idx[i] * dim + d];
    }
}

__global__ void k_sent_mean(const float* __restrict__ emb, const int* __restrict__ sw,
                            float* __restrict__ out, int ns, int dim, int L) {
    for (long long t = blockIdx.x * (long long)blockDim.x + threadIdx.x;
         t < (long long)ns * dim; t += (long long)gridDim.x * blockDim.x) {
        int s = (int)(t / dim), d = (int)(t % dim);
        float acc = 0.f;
        for (int w = 0; w < L; ++w) acc += emb[(size_t)sw[s * L + w] * dim + d];
        out[t] = acc / (float)L;
    }
}

__global__ void k_add_pos(const float* __restrict__ se, const float* __restrict__ pos,
                          const int* __restrict__ sp, float* __restrict__ out, int ns, int dim) {
    for (long long t = blockIdx.x * (long long)blockDim.x + threadIdx.x;
         t < (long long)ns * dim; t += (long long)gridDim.x * blockDim.x) {
        int s = (int)(t / dim), d = (int)(t % dim);
        out[t] = se[t] + pos[(size_t)sp[s] * dim + d];
    }
}

// out[i, :] = concat(a[i, 0:da], b[i/bdiv, 0:db])
__global__ void k_concat(const float* __restrict__ a, int da, const float* __restrict__ b,
                         int db, int bdiv, float* __restrict__ out, long long n) {
    int w = da + db;
    for (long long t = blockIdx.x * (long long)blockDim.x + threadIdx.x;
         t < n * w; t += (long long)gridDim.x * blockDim.x) {
        long long i = t / w; int j = (int)(t % w);
        out[t] = (j < da) ? a[i * da + j] : b[(i / bdiv) * db + (j - da)];
    }
}

__global__ void k_rowmean_seg(const float* __restrict__ X, float* __restrict__ out,
                              int bn, int seg, int dim) {
    for (int t = blockIdx.x * blockDim.x + threadIdx.x; t < bn * dim;
         t += gridDim.x * blockDim.x) {
        int b = t / dim, d = t % dim;
        float acc = 0.f;
        for (int s = 0; s < seg; ++s) acc += X[((size_t)b * seg + s) * dim + d];
        out[t] = acc / (float)seg;
    }
}

__global__ void k_node_score(const float* __restrict__ Z, const float* __restrict__ av,
                             float* __restrict__ sc, long long n, int heads, int dh) {
    for (long long t = blockIdx.x * (long long)blockDim.x + threadIdx.x;
         t < n * heads; t += (long long)gridDim.x * blockDim.x) {
        long long i = t / heads; int h = (int)(t % heads);
        float acc = 0.f;
        const float* z = Z + i * heads * dh + h * dh;
        const float* a = av + h * dh;
        for (int d = 0; d < dh; ++d) acc += z[d] * a[d];
        sc[t] = acc;
    }
}

// tab[b,h] = sum_f emb[b,f] * We[f,h]   (10 buckets, F=64)
__global__ void k_etab(const float* __restrict__ emb, const float* __restrict__ We,
                       float* __restrict__ tab, int heads) {
    int t = blockIdx.x * blockDim.x + threadIdx.x;
    if (t >= 10 * heads) return;
    int b = t / heads, h = t % heads;
    float acc = 0.f;
    for (int f = 0; f < FE; ++f) acc += emb[b * FE + f] * We[f * heads + h];
    tab[t] = acc;
}

__device__ inline void atomicMaxF(float* addr, float val) {
    int* ai = (int*)addr;
    int old_ = *ai, assumed;
    do {
        assumed = old_;
        if (__int_as_float(assumed) >= val) break;
        old_ = atomicCAS(ai, assumed, __float_as_int(val));
    } while (old_ != assumed);
}

__global__ void k_edge_score(const float* __restrict__ ssc, const float* __restrict__ dsc,
                             const float* __restrict__ etab, const int* __restrict__ esrc,
                             const int* __restrict__ edst, const int* __restrict__ eidx,
                             float* __restrict__ ebuf, float* __restrict__ dmax,
                             long long nE, int heads) {
    for (long long t = blockIdx.x * (long long)blockDim.x + threadIdx.x;
         t < nE * heads; t += (long long)gridDim.x * blockDim.x) {
        long long e = t / heads; int h = (int)(t % heads);
        float v = ssc[(size_t)esrc[e] * heads + h] + dsc[(size_t)edst[e] * heads + h];
        if (etab) v += etab[eidx[e] * heads + h];
        v = (v > 0.f) ? v : 0.2f * v;                 // leaky_relu(0.2)
        ebuf[t] = v;
        atomicMaxF(&dmax[(size_t)edst[e] * heads + h], v);
    }
}

__global__ void k_edge_exp(float* __restrict__ ebuf, const float* __restrict__ dmax,
                           float* __restrict__ dsum, const int* __restrict__ edst,
                           long long nE, int heads) {
    for (long long t = blockIdx.x * (long long)blockDim.x + threadIdx.x;
         t < nE * heads; t += (long long)gridDim.x * blockDim.x) {
        long long e = t / heads; int h = (int)(t % heads);
        float v = __expf(ebuf[t] - dmax[(size_t)edst[e] * heads + h]);
        ebuf[t] = v;
        atomicAdd(&dsum[(size_t)edst[e] * heads + h], v);
    }
}

__global__ void k_edge_agg(const float* __restrict__ ebuf, const float* __restrict__ dsum,
                           const float* __restrict__ Z, float* __restrict__ agg,
                           const int* __restrict__ esrc, const int* __restrict__ edst,
                           long long nE, int heads, int dh) {
    for (long long t = blockIdx.x * (long long)blockDim.x + threadIdx.x;
         t < nE * heads; t += (long long)gridDim.x * blockDim.x) {
        long long e = t / heads; int h = (int)(t % heads);
        int s = esrc[e], d = edst[e];
        float alpha = ebuf[t] / (dsum[(size_t)d * heads + h] + 1e-9f);
        const float* z = Z + (size_t)s * heads * dh + h * dh;
        float* o = agg + (size_t)d * heads * dh + h * dh;
        for (int k = 0; k < dh; ++k) atomicAdd(&o[k], z[k] * alpha);
    }
}

__global__ void k_elu_res(const float* __restrict__ agg, const float* __restrict__ hd,
                          float* __restrict__ out, long long n) {
    for (long long t = blockIdx.x * (long long)blockDim.x + threadIdx.x;
         t < n; t += (long long)gridDim.x * blockDim.x) {
        float x = agg[t];
        out[t] = ((x > 0.f) ? x : (__expf(x) - 1.f)) + hd[t];
    }
}

__global__ void k_gate(const float* __restrict__ G, const float* __restrict__ Cv,
                       const float* __restrict__ Rv, float* __restrict__ out, long long n) {
    for (long long t = blockIdx.x * (long long)blockDim.x + threadIdx.x;
         t < n; t += (long long)gridDim.x * blockDim.x) {
        float g = G[t];
        out[t] = g * Cv[t] + (1.f - g) * Rv[t];
    }
}

__global__ void k_layernorm(const float* __restrict__ X, const float* __restrict__ g,
                            const float* __restrict__ b, float* __restrict__ Y, int dim) {
    __shared__ float red[128];
    int row = blockIdx.x, d = threadIdx.x;
    float v = X[(size_t)row * dim + d];
    red[d] = v; __syncthreads();
    for (int s = dim / 2; s > 0; s >>= 1) { if (d < s) red[d] += red[d + s]; __syncthreads(); }
    float mu = red[0] / dim; __syncthreads();
    float dv = v - mu;
    red[d] = dv * dv; __syncthreads();
    for (int s = dim / 2; s > 0; s >>= 1) { if (d < s) red[d] += red[d + s]; __syncthreads(); }
    float var = red[0] / dim; __syncthreads();
    Y[(size_t)row * dim + d] = dv * rsqrtf(var + 1e-5f) * g[d] + b[d];
}

// =====================================================================
//  Sequential LSTM: one persistent block, h/c in LDS (torch gate order)
//  XW = x@Wih + b precomputed [B,S,4H]; Whh [H,4H]
// =====================================================================
__global__ __launch_bounds__(1024)
void k_lstm_seq(const float* __restrict__ XW, const float* __restrict__ Whh,
                float* __restrict__ Hout, int bn, int seq, int hh) {
    extern __shared__ float sm[];
    float* h = sm;
    float* c = sm + bn * hh;
    int total = bn * hh;
    for (int e = threadIdx.x; e < total; e += blockDim.x) { h[e] = 0.f; c[e] = 0.f; }
    __syncthreads();
    for (int t = 0; t < seq; ++t) {
        float nh[4], nc[4]; int cnt = 0;
        for (int e = threadIdx.x; e < total; e += blockDim.x) {
            int b = e / hh, u = e % hh;
            const float* xw = XW + ((size_t)b * seq + t) * (4 * hh);
            float gi = xw[u], gf = xw[hh + u], gg = xw[2 * hh + u], go = xw[3 * hh + u];
            const float* hb = h + b * hh;
            for (int k = 0; k < hh; ++k) {
                float hv = hb[k];
                const float* w = Whh + (size_t)k * 4 * hh;
                gi += hv * w[u]; gf += hv * w[hh + u];
                gg += hv * w[2 * hh + u]; go += hv * w[3 * hh + u];
            }
            float cc = c[e];
            float si = 1.f / (1.f + __expf(-gi));
            float sf = 1.f / (1.f + __expf(-gf));
            float so = 1.f / (1.f + __expf(-go));
            cc = sf * cc + si * tanhf(gg);
            nc[cnt] = cc; nh[cnt] = so * tanhf(cc); ++cnt;
        }
        __syncthreads();
        cnt = 0;
        for (int e = threadIdx.x; e < total; e += blockDim.x) {
            c[e] = nc[cnt]; h[e] = nh[cnt];
            int b = e / hh, u = e % hh;
            Hout[((size_t)b * seq + t) * hh + u] = nh[cnt]; ++cnt;
        }
        __syncthreads();
    }
}

// =====================================================================
//  MHA core (dh<=16): block per (b,h), thread per query row, 2-pass softmax
// =====================================================================
__global__ void k_mha(const float* __restrict__ Q, const float* __restrict__ K,
                      const float* __restrict__ V, float* __restrict__ O,
                      int seq, int heads, int dh) {
    int b = blockIdx.x / heads, h = blockIdx.x % heads;
    int q = threadIdx.x;
    if (q >= seq) return;
    int dm = heads * dh;
    const float* qp = Q + ((size_t)b * seq + q) * dm + h * dh;
    float qv[16];
    for (int d = 0; d < dh; ++d) qv[d] = qp[d];
    float scale = rsqrtf((float)dh);
    float m = -1e30f;
    for (int k = 0; k < seq; ++k) {
        const float* kp = K + ((size_t)b * seq + k) * dm + h * dh;
        float s = 0.f;
        for (int d = 0; d < dh; ++d) s += qv[d] * kp[d];
        m = fmaxf(m, s * scale);
    }
    float den = 0.f, acc[16];
    for (int d = 0; d < dh; ++d) acc[d] = 0.f;
    for (int k = 0; k < seq; ++k) {
        const float* kp = K + ((size_t)b * seq + k) * dm + h * dh;
        float s = 0.f;
        for (int d = 0; d < dh; ++d) s += qv[d] * kp[d];
        float e = __expf(s * scale - m);
        den += e;
        const float* vp = V + ((size_t)b * seq + k) * dm + h * dh;
        for (int d = 0; d < dh; ++d) acc[d] += e * vp[d];
    }
    float* op = O + ((size_t)b * seq + q) * dm + h * dh;
    for (int d = 0; d < dh; ++d) op[d] = acc[d] / den;
}

// =====================================================================
//  Host-side helpers
// =====================================================================
static void gemmL(hipStream_t st, const float* A, const float* B, const float* bias,
                  const float* res, float* C, int M, int N, int K, int act) {
    dim3 g(cdiv(M, 16), cdiv(N, 32));
    k_gemm_wmma<<<g, dim3(32), 0, st>>>(A, B, bias, res, C, M, N, K, act);
}
static unsigned blocks_for(long long n) {
    long long b = (n + 255) / 256;
    if (b > (1 << 20)) b = 1 << 20;
    if (b < 1) b = 1;
    return (unsigned)b;
}
static void fillL(hipStream_t st, float* p, long long n, float v) {
    k_fill<<<blocks_for(n), 256, 0, st>>>(p, n, v);
}

struct GatBufs {
    float *Z, *ZD, *SSC, *DSC, *ETAB, *EBUF, *DMAX, *DSUM, *AGG, *H1, *T;
};

static void run_gat(hipStream_t st, const GatBufs& w,
                    const float* hs, int n_src, int d_src,
                    const float* hd, int n_dst, int d_dst,
                    const int* esrc, const int* edst, long long nE,
                    const int* eidx, const float* etable,
                    const float* W, const float* Wd, const float* as_, const float* ad_,
                    const float* We, const float* W1, const float* W2,
                    int heads, int dh, float* out) {
    const int hdm = heads * dh;   // == d_dst
    gemmL(st, hs, W,  nullptr, nullptr, w.Z,  n_src, hdm, d_src, 0);
    gemmL(st, hd, Wd, nullptr, nullptr, w.ZD, n_dst, hdm, d_dst, 0);
    k_node_score<<<blocks_for((long long)n_src * heads), 256, 0, st>>>(w.Z, as_, w.SSC, n_src, heads, dh);
    k_node_score<<<blocks_for((long long)n_dst * heads), 256, 0, st>>>(w.ZD, ad_, w.DSC, n_dst, heads, dh);
    const float* etab = nullptr;
    if (eidx) {
        k_etab<<<1, 128, 0, st>>>(etable, We, w.ETAB, heads);
        etab = w.ETAB;
    }
    fillL(st, w.DMAX, (long long)n_dst * heads, -1e30f);
    unsigned nb = blocks_for(nE * heads);
    k_edge_score<<<nb, 256, 0, st>>>(w.SSC, w.DSC, etab, esrc, edst, eidx,
                                     w.EBUF, w.DMAX, nE, heads);
    fillL(st, w.DSUM, (long long)n_dst * heads, 0.f);
    k_edge_exp<<<nb, 256, 0, st>>>(w.EBUF, w.DMAX, w.DSUM, edst, nE, heads);
    fillL(st, w.AGG, (long long)n_dst * hdm, 0.f);
    k_edge_agg<<<nb, 256, 0, st>>>(w.EBUF, w.DSUM, w.Z, w.AGG, esrc, edst, nE, heads, dh);
    k_elu_res<<<blocks_for((long long)n_dst * hdm), 256, 0, st>>>(w.AGG, hd, w.H1,
                                                                 (long long)n_dst * hdm);
    // FFN with residual, chunked so the 512-wide intermediate stays small
    const int CH = 8192;
    for (int r0 = 0; r0 < n_dst; r0 += CH) {
        int m = (n_dst - r0 < CH) ? (n_dst - r0) : CH;
        gemmL(st, w.H1 + (size_t)r0 * hdm, W1, nullptr, nullptr, w.T, m, FFN, hdm, 1);
        gemmL(st, w.T, W2, nullptr, w.H1 + (size_t)r0 * hdm,
              out + (size_t)r0 * hdm, m, hdm, FFN, 0);
    }
}

// =====================================================================
extern "C" void kernel_launch(void* const* d_in, const int* in_sizes, int n_in,
                              void* d_out, int out_size, void* d_ws, size_t ws_size,
                              hipStream_t stream) {
    (void)in_sizes; (void)n_in; (void)out_size; (void)ws_size;
    hipStream_t st = stream;
    auto F = [&](int i) { return (const float*)d_in[i]; };
    auto I = [&](int i) { return (const int*)d_in[i]; };

    // params flattened in setup_inputs() insertion order
    const float *embed = F(0), *tf_embed = F(1), *sim_embed = F(2), *pos_table = F(3);
    const float *cnn_W = F(4), *cnn_b = F(5);
    const float *l1_Wih = F(6), *l1_Whh = F(7), *l1_b = F(8);
    const float *lp_W = F(9), *lp_b = F(10);
    const float *tr_Wq = F(11), *tr_Wk = F(12), *tr_Wv = F(13), *tr_Wo = F(14);
    const float *trp_W = F(15), *trp_b = F(16);
    const float *nfp_W = F(17), *dn_W = F(18);
    const float *w2s_W = F(19), *w2s_Wd = F(20), *w2s_as = F(21), *w2s_ad = F(22),
                *w2s_We = F(23), *w2s_W1 = F(24), *w2s_W2 = F(25);
    const float *w2w_W = F(26), *w2w_Wd = F(27), *w2w_as = F(28), *w2w_ad = F(29),
                *w2w_W1 = F(30), *w2w_W2 = F(31);
    const float *s2w_W = F(32), *s2w_Wd = F(33), *s2w_as = F(34), *s2w_ad = F(35),
                *s2w_We = F(36), *s2w_W1 = F(37), *s2w_W2 = F(38);
    const float *s2s_W = F(39), *s2s_Wd = F(40), *s2s_as = F(41), *s2s_ad = F(42),
                *s2s_We = F(43), *s2s_W1 = F(44), *s2s_W2 = F(45);
    const float *dm_Wq = F(46), *dm_Wk = F(47), *dm_Wv = F(48), *dm_Wo = F(49);
    const float *Wc = F(50), *Wr = F(51), *fg = F(52);
    const float *ln_g = F(53), *ln_b = F(54);
    const float *lo_Wih = F(55), *lo_Whh = F(56), *lo_b = F(57);
    const float *wh_W = F(58), *wh_b = F(59);
    const int *wid = I(60), *sent_words = I(61), *sent_pos = I(62);
    const int *e_ws_src = I(63), *e_ws_dst = I(64), *e_ws_tf = I(65);
    const int *e_ww_src = I(66), *e_ww_dst = I(67);
    const int *e_ss_src = I(68), *e_ss_dst = I(69), *e_ss_sim = I(70);
    float* out = (float*)d_out;

    // ---- deterministic scratch layout ----
    size_t off = 0;
    auto A = [&](size_t n) { float* p = (float*)d_ws + off; off += n; return p; };
    float* WS_A = A((size_t)NW * VD);          // word state ping
    float* WS_B = A((size_t)NW * VD);          // word state pong
    GatBufs g;
    g.Z    = A((size_t)NW * VD);
    g.ZD   = A((size_t)NW * VD);
    g.SSC  = A((size_t)NW * 8);
    g.DSC  = A((size_t)NW * 8);
    g.ETAB = A(128);
    g.EBUF = A((size_t)EWS * 8);
    g.DMAX = A((size_t)NW * 8);
    g.DSUM = A((size_t)NW * 8);
    g.AGG  = A((size_t)NW * VD);
    g.H1   = A((size_t)NW * VD);
    g.T    = A((size_t)8192 * FFN);
    float* SE    = A((size_t)NS * VD);
    float* CNNIN = A((size_t)NS * VD);
    float* CNN   = A((size_t)NS * NF);
    float* XW1   = A((size_t)NS * 4 * NF);
    float* LS1   = A((size_t)NS * NF);
    float* LF    = A((size_t)NS * NF);
    float* Qb    = A((size_t)NS * 128);
    float* Kb    = A((size_t)NS * 128);
    float* Vb    = A((size_t)NS * 128);
    float* AT    = A((size_t)NS * 128);
    float* AO    = A((size_t)NS * 128);
    float* TR    = A((size_t)NS * NF);
    float* NODE  = A((size_t)NS * 2 * NF);
    float* SF    = A((size_t)NS * HH);
    float* DOCM  = A((size_t)BB * HH);
    float* DOC   = A((size_t)BB * HH);
    float* SS_A  = A((size_t)NS * HH);
    float* SS_B  = A((size_t)NS * HH);
    float* S2    = A((size_t)NS * 2 * HH);
    float* Cb    = A((size_t)NS * 2 * HH);
    float* Rb    = A((size_t)NS * 2 * HH);
    float* CR    = A((size_t)NS * 4 * HH);
    float* Gb    = A((size_t)NS * 2 * HH);
    float* HG    = A((size_t)NS * 2 * HH);
    float* HN    = A((size_t)NS * 2 * HH);
    float* XW2   = A((size_t)NS * 8 * HH);
    float* O2    = A((size_t)NS * 2 * HH);

    // ================= Stage A: node features =================
    k_gather_rows<<<blocks_for((long long)NW * VD), 256, 0, st>>>(embed, wid, WS_A, NW, VD);
    k_sent_mean<<<blocks_for((long long)NS * VD), 256, 0, st>>>(embed, sent_words, SE, NS, VD, 32);
    k_add_pos<<<blocks_for((long long)NS * VD), 256, 0, st>>>(SE, pos_table, sent_pos, CNNIN, NS, VD);
    gemmL(st, CNNIN, cnn_W, cnn_b, nullptr, CNN, NS, NF, VD, 0);
    gemmL(st, SE, l1_Wih, l1_b, nullptr, XW1, NS, 4 * NF, VD, 0);
    k_lstm_seq<<<1, 1024, 2 * BB * NF * sizeof(float), st>>>(XW1, l1_Whh, LS1, BB, SS_, NF);
    gemmL(st, LS1, lp_W, lp_b, nullptr, LF, NS, NF, NF, 0);
    // sentence-level transformer (4 heads, dh=16)
    gemmL(st, LF, tr_Wq, nullptr, nullptr, Qb, NS, NF, NF, 0);
    gemmL(st, LF, tr_Wk, nullptr, nullptr, Kb, NS, NF, NF, 0);
    gemmL(st, LF, tr_Wv, nullptr, nullptr, Vb, NS, NF, NF, 0);
    k_mha<<<BB * 4, SS_, 0, st>>>(Qb, Kb, Vb, AT, SS_, 4, 16);
    gemmL(st, AT, tr_Wo, nullptr, LF, AO, NS, NF, NF, 0);              // lf + mha
    gemmL(st, AO, trp_W, trp_b, nullptr, TR, NS, NF, NF, 0);
    k_concat<<<blocks_for((long long)NS * 2 * NF), 256, 0, st>>>(CNN, NF, TR, NF, 1, NODE, NS);
    gemmL(st, NODE, nfp_W, nullptr, nullptr, SF, NS, HH, 2 * NF, 0);
    k_rowmean_seg<<<blocks_for(BB * HH), 256, 0, st>>>(SF, DOCM, BB, SS_, HH);
    gemmL(st, DOCM, dn_W, nullptr, nullptr, DOC, BB, HH, HH, 0);

    // ================= Stage B: WSWGAT message passing =================
    // ss = gat_w2s(word_feat -> sent_feat)
    run_gat(st, g, WS_A, NW, VD, SF, NS, HH, e_ws_src, e_ws_dst, EWS,
            e_ws_tf, tf_embed, w2s_W, w2s_Wd, w2s_as, w2s_ad, w2s_We,
            w2s_W1, w2s_W2, 8, 8, SS_A);
    // NITER = 1
    run_gat(st, g, WS_A, NW, VD, WS_A, NW, VD, e_ww_src, e_ww_dst, EWW,
            nullptr, nullptr, w2w_W, w2w_Wd, w2w_as, w2w_ad, nullptr,
            w2w_W1, w2w_W2, 6, 20, WS_B);                               // w2w
    run_gat(st, g, SS_A, NS, HH, WS_B, NW, VD, e_ws_dst, e_ws_src, EWS,
            e_ws_tf, tf_embed, s2w_W, s2w_Wd, s2w_as, s2w_ad, s2w_We,
            s2w_W1, s2w_W2, 6, 20, WS_A);                               // s2w
    run_gat(st, g, WS_A, NW, VD, SS_A, NS, HH, e_ws_src, e_ws_dst, EWS,
            e_ws_tf, tf_embed, w2s_W, w2s_Wd, w2s_as, w2s_ad, w2s_We,
            w2s_W1, w2s_W2, 8, 8, SS_B);                                // w2s
    run_gat(st, g, SS_B, NS, HH, SS_B, NS, HH, e_ss_src, e_ss_dst, ESS,
            e_ss_sim, sim_embed, s2s_W, s2s_Wd, s2s_as, s2s_ad, s2s_We,
            s2s_W1, s2s_W2, 8, 8, SS_A);                                // s2s

    // ================= Stage C: doc-aware fusion + readout =================
    k_concat<<<blocks_for((long long)NS * 2 * HH), 256, 0, st>>>(SS_A, HH, DOC, HH, SS_, S2, NS);
    gemmL(st, S2, dm_Wq, nullptr, nullptr, Qb, NS, 2 * HH, 2 * HH, 0);
    gemmL(st, S2, dm_Wk, nullptr, nullptr, Kb, NS, 2 * HH, 2 * HH, 0);
    gemmL(st, S2, dm_Wv, nullptr, nullptr, Vb, NS, 2 * HH, 2 * HH, 0);
    k_mha<<<BB * 8, SS_, 0, st>>>(Qb, Kb, Vb, AT, SS_, 8, 16);
    gemmL(st, AT, dm_Wo, nullptr, nullptr, AO, NS, 2 * HH, 2 * HH, 0);
    gemmL(st, AO, Wc, nullptr, nullptr, Cb, NS, 2 * HH, 2 * HH, 0);
    gemmL(st, S2, Wr, nullptr, nullptr, Rb, NS, 2 * HH, 2 * HH, 0);
    k_concat<<<blocks_for((long long)NS * 4 * HH), 256, 0, st>>>(Cb, 2 * HH, Rb, 2 * HH, 1, CR, NS);
    gemmL(st, CR, fg, nullptr, nullptr, Gb, NS, 2 * HH, 4 * HH, 2);     // sigmoid gate
    k_gate<<<blocks_for((long long)NS * 2 * HH), 256, 0, st>>>(Gb, Cb, Rb, HG,
                                                               (long long)NS * 2 * HH);
    k_layernorm<<<NS, 2 * HH, 0, st>>>(HG, ln_g, ln_b, HN, 2 * HH);
    gemmL(st, HN, lo_Wih, lo_b, nullptr, XW2, NS, 8 * HH, 2 * HH, 0);
    k_lstm_seq<<<1, 1024, 2 * BB * 2 * HH * sizeof(float), st>>>(XW2, lo_Whh, O2, BB, SS_, 2 * HH);
    gemmL(st, O2, wh_W, wh_b, nullptr, out, NS, 2, 2 * HH, 2);          // sigmoid head
}
